// JambaRLPolicy_53919019434500
// MI455X (gfx1250) — matrix operand order repaired and checked by reading, not compiled, not run on hardware
//
#include <hip/hip_runtime.h>
#include <math.h>

// ---------------------------------------------------------------------------
// JambaRLPolicy forward, specialized for L==1 (attention == wv->wo, mamba scan
// collapses to a per-row scalar). All heavy lifting is a bf16 WMMA tiled GEMM
// (v_wmma_f32_16x16x32_bf16) with LDS double-buffering and fused epilogues.
// ---------------------------------------------------------------------------

typedef __attribute__((ext_vector_type(16))) __bf16 v16bf;
typedef __attribute__((ext_vector_type(8)))  __bf16 bf16x8;
typedef __attribute__((ext_vector_type(4)))  __bf16 bf16x4;
typedef __attribute__((ext_vector_type(2)))  __bf16 bf16x2;
typedef __attribute__((ext_vector_type(8)))  float  v8f;

enum EpiMode { EPI_STORE = 0, EPI_ADD_RES, EPI_TANH, EPI_SOFTPLUS, EPI_SILU, EPI_MUL };

#define BM 128
#define BN 128
#define BK 32
#define BKP 40   // padded LDS stride in bf16 elems (80 B, keeps 16B alignment)

__device__ __forceinline__ float siluf(float v)     { return v / (1.f + __expf(-v)); }
__device__ __forceinline__ float softplus_f(float v){ return v > 20.f ? v : log1pf(__expf(v)); }

// ---------------------------------------------------------------------------
// Tiled GEMM: C(M x N) = epilogue(A(M x K, lda) @ B(K x N) + bias)
// A, B are f32 in HBM; converted to bf16 while staging into LDS (weights are
// read once in f32, never round-tripped). 256 threads = 8 wave32 waves; wave w
// covers rows [(w&1)*64, +64), cols [(w>>1)*32, +32) of the 128x128 block tile
// -> 4x2 grid of 16x16x32 WMMA accumulators. Double-buffered LDS: one barrier
// per K step. Stagers issue ALL global loads before any wait (explicit
// load-phase arrays) so HBM latency overlaps the 8 WMMAs of the current tile.
// M % 128 == 0 and K % 32 == 0 hold for every call here. FULLN => N % 128 == 0.
// ---------------------------------------------------------------------------
template <int MODE, bool FULLN>
__global__ __launch_bounds__(256)
void gemm_bf16_wmma(const float* __restrict__ A, int lda,
                    const float* __restrict__ Bw,
                    const float* __restrict__ bias,   // nullable
                    const float* __restrict__ aux,    // residual / mul partner (ld = N)
                    float* __restrict__ C,
                    int N, int K)
{
    __shared__ __bf16 sA[2][BM * BKP];
    __shared__ __bf16 sB[2][BN * BKP];

    const int tid     = threadIdx.x;
    const int rowBase = blockIdx.y * BM;
    const int colBase = blockIdx.x * BN;
    const int wave    = tid >> 5;
    const int lane    = tid & 31;
    const int wm0     = (wave & 1) * 64;   // wave row offset inside tile
    const int wn0     = (wave >> 1) * 32;  // wave col offset inside tile
    const int lhalf   = lane >> 4;         // 0 / 1
    const int lmod    = lane & 15;

    // ---- hoisted per-thread staging offsets (K-invariant) ----
    size_t aOff[4];  int sAoff[4];
#pragma unroll
    for (int i = 0; i < 4; ++i) {
        int idx  = tid + i * 256;          // 0..1023 float4 groups
        int r    = idx >> 3;               // 0..127
        int c    = (idx & 7) << 2;         // 0,4,...,28
        aOff[i]  = (size_t)(rowBase + r) * lda + c;
        sAoff[i] = r * BKP + c;
    }
    size_t bOff[8];  int sBoff[8];  bool bOk[8];
#pragma unroll
    for (int i = 0; i < 8; ++i) {
        int idx  = tid + i * 256;          // 0..2047 k-pair slots
        int kp   = idx >> 7;               // 0..15
        int n    = idx & 127;
        int gn   = colBase + n;
        bOk[i]   = FULLN || (gn < N);
        bOff[i]  = (size_t)(kp * 2) * N + gn;
        sBoff[i] = n * BKP + kp * 2;
    }

    // ---- tile stagers: load phase first (all loads in flight), then store ----
    auto stageA = [&](int k0, int buf) {
        float4 v[4];
#pragma unroll
        for (int i = 0; i < 4; ++i)
            v[i] = *(const float4*)(A + aOff[i] + k0);
#pragma unroll
        for (int i = 0; i < 4; ++i) {
            bf16x4 bv;
            bv[0] = (__bf16)v[i].x; bv[1] = (__bf16)v[i].y;
            bv[2] = (__bf16)v[i].z; bv[3] = (__bf16)v[i].w;
            *(bf16x4*)&sA[buf][sAoff[i]] = bv;
        }
    };
    auto stageB = [&](int k0, int buf) {   // stored transposed: sB[n][k]
        const float* Bk = Bw + (size_t)k0 * N;
        float w0[8], w1[8];
#pragma unroll
        for (int i = 0; i < 8; ++i) {
            if (bOk[i]) {
                w0[i] = Bk[bOff[i]];
                w1[i] = Bk[bOff[i] + N];
            } else { w0[i] = 0.f; w1[i] = 0.f; }
        }
#pragma unroll
        for (int i = 0; i < 8; ++i) {
            bf16x2 p; p[0] = (__bf16)w0[i]; p[1] = (__bf16)w1[i];
            *(bf16x2*)&sB[buf][sBoff[i]] = p;
        }
    };

    v8f acc[4][2];
#pragma unroll
    for (int i = 0; i < 4; ++i)
#pragma unroll
        for (int j = 0; j < 2; ++j)
#pragma unroll
            for (int e = 0; e < 8; ++e) acc[i][j][e] = 0.f;

    const int nk = K / BK;
    stageA(0, 0);
    stageB(0, 0);
    __syncthreads();

    for (int kt = 0; kt < nk; ++kt) {
        const int cur = kt & 1;
        if (kt + 1 < nk) {                 // overlap next tile with compute
            stageA((kt + 1) * BK, cur ^ 1);
            stageB((kt + 1) * BK, cur ^ 1);
        }

        // ---- fragments (layouts per CDNA5 ISA 7.12.2) ----
        v16bf afrag[4];
#pragma unroll
        for (int wm = 0; wm < 4; ++wm) {
            int r = wm0 + wm * 16 + lmod;
            bf16x8 lo = *(const bf16x8*)&sA[cur][r * BKP + lhalf * 8];
            bf16x8 hi = *(const bf16x8*)&sA[cur][r * BKP + 16 + lhalf * 8];
            afrag[wm] = __builtin_shufflevector(lo, hi, 0,1,2,3,4,5,6,7,8,9,10,11,12,13,14,15);
        }
        v16bf bfrag[2];
#pragma unroll
        for (int wn = 0; wn < 2; ++wn) {
            int n = wn0 + wn * 16 + lmod;
            bf16x8 lo = *(const bf16x8*)&sB[cur][n * BKP + lhalf * 8];
            bf16x8 hi = *(const bf16x8*)&sB[cur][n * BKP + 16 + lhalf * 8];
            bfrag[wn] = __builtin_shufflevector(lo, hi, 0,1,2,3,4,5,6,7,8,9,10,11,12,13,14,15);
        }
#pragma unroll
        for (int wm = 0; wm < 4; ++wm)
#pragma unroll
            for (int wn = 0; wn < 2; ++wn)
                acc[wm][wn] = __builtin_amdgcn_wmma_f32_16x16x32_bf16(
                    false, afrag[wm], false, bfrag[wn],
                    (short)0, acc[wm][wn], false, false);
        __syncthreads();   // single barrier per K step (double-buffered)
    }

    // ---- epilogue ----
#pragma unroll
    for (int wm = 0; wm < 4; ++wm) {
#pragma unroll
        for (int wn = 0; wn < 2; ++wn) {
            int col = colBase + wn0 + wn * 16 + lmod;
            if (FULLN || col < N) {
                float bval = bias ? bias[col] : 0.f;
#pragma unroll
                for (int r = 0; r < 8; ++r) {
                    int row  = rowBase + wm0 + wm * 16 + lhalf * 8 + r;
                    size_t o = (size_t)row * N + col;
                    float v  = acc[wm][wn][r] + bval;
                    if (MODE == EPI_STORE)         C[o] = v;
                    else if (MODE == EPI_ADD_RES)  C[o] = aux[o] + v;
                    else if (MODE == EPI_TANH)     C[o] = tanhf(v);
                    else if (MODE == EPI_SOFTPLUS) C[o] = softplus_f(v);
                    else if (MODE == EPI_SILU)     C[o] = siluf(v);
                    else if (MODE == EPI_MUL)      C[o] = v * aux[o];
                }
            }
        }
    }
}

// ---------------------------------------------------------------------------
// LayerNorm over D=1024, one block per row, optional fused tanh.
// ---------------------------------------------------------------------------
__global__ __launch_bounds__(256)
void ln_kernel(const float* __restrict__ in, const float* __restrict__ g,
               const float* __restrict__ b, float* __restrict__ out, int do_tanh)
{
    const int D_ = 1024;
    __shared__ float sred[16];
    int row = blockIdx.x;
    const float* x = in + (size_t)row * D_;
    float vals[4], s = 0.f, s2 = 0.f;
#pragma unroll
    for (int j = 0; j < 4; ++j) {
        float v = x[threadIdx.x + j * 256];
        vals[j] = v; s += v; s2 += v * v;
    }
#pragma unroll
    for (int off = 16; off > 0; off >>= 1) {
        s  += __shfl_down(s,  off);
        s2 += __shfl_down(s2, off);
    }
    int wv = threadIdx.x >> 5, lane = threadIdx.x & 31;
    if (lane == 0) { sred[wv] = s; sred[8 + wv] = s2; }
    __syncthreads();
    if (threadIdx.x == 0) {
        float ts = 0.f, ts2 = 0.f;
        for (int w = 0; w < 8; ++w) { ts += sred[w]; ts2 += sred[8 + w]; }
        sred[0] = ts; sred[8] = ts2;
    }
    __syncthreads();
    float mean = sred[0] / D_;
    float var  = sred[8] / D_ - mean * mean;
    float rstd = rsqrtf(var + 1e-5f);
#pragma unroll
    for (int j = 0; j < 4; ++j) {
        int i   = threadIdx.x + j * 256;
        float v = (vals[j] - mean) * rstd * g[i] + b[i];
        out[(size_t)row * D_ + i] = do_tanh ? tanhf(v) : v;
    }
}

// xc = silu(xi * conv_w[:, -1] + conv_b); xi = xz[:, :2048], xz ld = 4096
__global__ __launch_bounds__(256)
void mamba_pre(const float* __restrict__ xz, const float* __restrict__ conv_w,
               const float* __restrict__ conv_b, float* __restrict__ xc)
{
    int i   = blockIdx.x * 256 + threadIdx.x;   // 0 .. 1024*2048-1
    int row = i >> 11;
    int d   = i & 2047;
    float v = xz[(size_t)row * 4096 + d] * conv_w[d * 4 + 3] + conv_b[d];
    xc[i] = siluf(v);
}

// u = (delta*xc*dot(Bm,Cm) + xc*Dp) * silu(z);  one block per row
__global__ __launch_bounds__(256)
void mamba_scan(const float* __restrict__ delta, const float* __restrict__ xc,
                const float* __restrict__ dbl, const float* __restrict__ xz,
                const float* __restrict__ Dp, float* __restrict__ u)
{
    __shared__ float sbc;
    int row = blockIdx.x;
    if (threadIdx.x == 0) {
        float bc = 0.f;
        const float* r = dbl + (size_t)row * 96;
#pragma unroll
        for (int s = 0; s < 16; ++s) bc += r[64 + s] * r[80 + s];
        sbc = bc;
    }
    __syncthreads();
    float bc = sbc;
#pragma unroll
    for (int j = 0; j < 8; ++j) {
        int d      = threadIdx.x + j * 256;
        size_t o   = (size_t)row * 2048 + d;
        float xcv  = xc[o];
        float y    = delta[o] * xcv * bc + xcv * Dp[d];
        float z    = xz[(size_t)row * 4096 + 2048 + d];
        u[o] = y * siluf(z);
    }
}

// ---------------------------------------------------------------------------
// Host orchestration
// ---------------------------------------------------------------------------
template <int MODE>
static void gemm_launch(hipStream_t s, const float* A, int lda, const float* B,
                        const float* bias, const float* aux, float* C, int N, int K)
{
    dim3 grid((N + BN - 1) / BN, 1024 / BM);
    dim3 blk(256);
    if ((N % BN) == 0)
        gemm_bf16_wmma<MODE, true ><<<grid, blk, 0, s>>>(A, lda, B, bias, aux, C, N, K);
    else
        gemm_bf16_wmma<MODE, false><<<grid, blk, 0, s>>>(A, lda, B, bias, aux, C, N, K);
}

static void gemm(hipStream_t s, int mode, const float* A, int lda, const float* B,
                 const float* bias, const float* aux, float* C, int N, int K)
{
    switch (mode) {
    case EPI_STORE:    gemm_launch<EPI_STORE   >(s, A, lda, B, bias, aux, C, N, K); break;
    case EPI_ADD_RES:  gemm_launch<EPI_ADD_RES >(s, A, lda, B, bias, aux, C, N, K); break;
    case EPI_TANH:     gemm_launch<EPI_TANH    >(s, A, lda, B, bias, aux, C, N, K); break;
    case EPI_SOFTPLUS: gemm_launch<EPI_SOFTPLUS>(s, A, lda, B, bias, aux, C, N, K); break;
    case EPI_SILU:     gemm_launch<EPI_SILU    >(s, A, lda, B, bias, aux, C, N, K); break;
    case EPI_MUL:      gemm_launch<EPI_MUL     >(s, A, lda, B, bias, aux, C, N, K); break;
    }
}

extern "C" void kernel_launch(void* const* d_in, const int* in_sizes, int n_in,
                              void* d_out, int out_size, void* d_ws, size_t ws_size,
                              hipStream_t stream)
{
    int idx = 0;
    auto F = [&]() { return (const float*)d_in[idx++]; };

    const float* obs     = F();
    const float* in_w    = F();  const float* in_b    = F();
    const float* in_ln_g = F();  const float* in_ln_b = F();

    struct LayerP {
        const float *ln_g, *ln_b, *ln2_g, *ln2_b, *gate_w, *up_w, *down_w;
        bool attn;
        const float *wq, *wk, *wv, *wo;                                       // attn
        const float *m_in_w, *conv_w, *conv_b, *x_proj_w, *dt_w, *dt_b,       // mamba
                    *A_log, *Dp, *out_w, *out_b;
    } L[8];

    for (int i = 0; i < 8; ++i) {
        L[i].ln_g = F(); L[i].ln_b = F(); L[i].ln2_g = F(); L[i].ln2_b = F();
        L[i].gate_w = F(); L[i].up_w = F(); L[i].down_w = F();
        L[i].attn = ((i + 1) % 4 == 0);
        if (L[i].attn) {
            L[i].wq = F(); L[i].wk = F(); L[i].wv = F(); L[i].wo = F();
        } else {
            L[i].m_in_w = F(); L[i].conv_w = F(); L[i].conv_b = F();
            L[i].x_proj_w = F(); L[i].dt_w = F(); L[i].dt_b = F();
            L[i].A_log = F(); L[i].Dp = F(); L[i].out_w = F(); L[i].out_b = F();
        }
    }
    const float* f_ln_g = F(); const float* f_ln_b = F();
    const float* o1_w   = F(); const float* o1_b   = F();
    const float* o2_w   = F(); const float* o2_b   = F();

    // workspace layout (floats)
    float* ws = (float*)d_ws;
    float* H  = ws;  ws += 1024 * 1024;   // residual stream
    float* X  = ws;  ws += 1024 * 1024;   // LN output
    float* T0 = ws;  ws += 1024 * 4096;   // xz / silu(gate)
    float* T1 = ws;  ws += 1024 * 4096;   // gated MLP activation
    float* T2 = ws;  ws += 1024 * 2048;   // xc -> u
    float* T3 = ws;  ws += 1024 * 2048;   // delta
    float* T4 = ws;  ws += 1024 * 96;     // dbl (dt|Bm|Cm)
    float* T5 = ws;  ws += 1024 * 1024;   // v / o1

    // ---- input embed: tanh(ln(obs @ in_w + in_b)) ----
    gemm(stream, EPI_STORE, obs, 512, in_w, in_b, nullptr, T5, 1024, 512);
    ln_kernel<<<1024, 256, 0, stream>>>(T5, in_ln_g, in_ln_b, H, 1);

    for (int i = 0; i < 8; ++i) {
        const LayerP& lp = L[i];
        // mixer
        ln_kernel<<<1024, 256, 0, stream>>>(H, lp.ln_g, lp.ln_b, X, 0);
        if (lp.attn) {
            // L==1: softmax(1x1)==1 -> y = (x @ wv) @ wo (wq, wk dead)
            gemm(stream, EPI_STORE,   X,  1024, lp.wv, nullptr, nullptr, T5, 1024, 1024);
            gemm(stream, EPI_ADD_RES, T5, 1024, lp.wo, nullptr, H,       H,  1024, 1024);
        } else {
            gemm(stream, EPI_STORE, X, 1024, lp.m_in_w, nullptr, nullptr, T0, 4096, 1024);
            mamba_pre<<<(1024 * 2048) / 256, 256, 0, stream>>>(T0, lp.conv_w, lp.conv_b, T2);
            gemm(stream, EPI_STORE,    T2, 2048, lp.x_proj_w, nullptr, nullptr, T4, 96,   2048);
            gemm(stream, EPI_SOFTPLUS, T4, 96,   lp.dt_w,     lp.dt_b, nullptr, T3, 2048, 64);
            mamba_scan<<<1024, 256, 0, stream>>>(T3, T2, T4, T0, lp.Dp, T2);
            gemm(stream, EPI_ADD_RES, T2, 2048, lp.out_w, lp.out_b, H, H, 1024, 2048);
        }
        // MLP
        ln_kernel<<<1024, 256, 0, stream>>>(H, lp.ln2_g, lp.ln2_b, X, 0);
        gemm(stream, EPI_SILU,    X,  1024, lp.gate_w, nullptr, nullptr, T0, 4096, 1024);
        gemm(stream, EPI_MUL,     X,  1024, lp.up_w,   nullptr, T0,      T1, 4096, 1024);
        gemm(stream, EPI_ADD_RES, T1, 4096, lp.down_w, nullptr, H,       H,  1024, 4096);
    }

    // ---- head ----
    ln_kernel<<<1024, 256, 0, stream>>>(H, f_ln_g, f_ln_b, X, 0);
    gemm(stream, EPI_TANH, X,  1024, o1_w, o1_b, nullptr, T5, 1024, 1024);
    gemm(stream, EPI_TANH, T5, 1024, o2_w, o2_b, nullptr, (float*)d_out, 64, 1024);
}
